// MultiHeadAttention_39642548142085
// MI455X (gfx1250) — matrix operand order repaired and checked by reading, compiled, not run on hardware
//
#include <hip/hip_runtime.h>

// MI455X (gfx1250) multi-head causal attention, bf16 WMMA + fp32 accum.
// Pipeline: cvt(x,W) -> QKV GEMM -> flash-attention -> output projection.
// GEMM K-loops are software-pipelined so s_wait_loadcnt stages instead of
// serializing each WMMA behind its own loads.

typedef unsigned short u16;
typedef __attribute__((ext_vector_type(16))) __bf16 v16bf;
typedef __attribute__((ext_vector_type(8)))  __bf16 v8bf;
typedef __attribute__((ext_vector_type(8)))  float  v8f;

namespace {
constexpr int    kB = 4, kT = 2048, kC = 1024, kH = 16, kD = 64;
constexpr size_t XN = (size_t)kB * kT * kC;        // 8,388,608 (x elements)
constexpr size_t WN = (size_t)kH * kC * kD;        // 1,048,576 (one proj weight; == Wo size)
constexpr size_t QN = (size_t)kB * kH * kT * kD;   // 8,388,608 (q/k/v elements)
}

// ---------------- helpers ----------------

static __device__ __forceinline__ u16 f2bf(float f) {
  union { float f; unsigned u; } cv; cv.f = f;
  unsigned u = cv.u + 0x7FFFu + ((cv.u >> 16) & 1u);  // round-to-nearest-even
  return (u16)(u >> 16);
}

// A-fragment (16x32 bf16, wave32): lane = row (lane&15); kbase = lane<16 ? 0 : 16.
// half e -> K = kbase + e. Two contiguous 16B loads.
static __device__ __forceinline__ v16bf load_afrag(const u16* p, int lane) {
  const int kb = (lane < 16) ? 0 : 16;
  v8bf lo = *(const v8bf*)(p + kb);
  v8bf hi = *(const v8bf*)(p + kb + 8);
  v16bf r;
#pragma unroll
  for (int i = 0; i < 8; ++i) { r[i] = lo[i]; r[i + 8] = hi[i]; }
  return r;
}

// B-fragment (32x16 bf16, wave32): lane = column N (lane&15); half e -> K = kbase + e,
// kbase = lane<16 ? 0 : 16. One contiguous 32B load from an N-major row.
static __device__ __forceinline__ v16bf load_bfrag(const u16* p, int lane) {
  const int kb = (lane < 16) ? 0 : 16;
  return *(const v16bf*)(p + kb);
}

static __device__ __forceinline__ v8f wmma_bf16(v16bf a, v16bf b, v8f c) {
  // (neg_a, A, neg_b, B, c_mod, C, reuse_a, reuse_b)
  return __builtin_amdgcn_wmma_f32_16x16x32_bf16(false, a, false, b, (short)0, c,
                                                 false, false);
}

// ---------------- conversion kernels ----------------

__global__ __launch_bounds__(256) void cvt_x_kernel(const float* __restrict__ in,
                                                    u16* __restrict__ o, int n) {
  int i = blockIdx.x * 256 + threadIdx.x;
  if (i < n) o[i] = f2bf(in[i]);
}

// W [H][C][D] fp32 -> Wt [H][D][C] bf16  (N-major for B-fragments)
__global__ __launch_bounds__(256) void cvt_w_kernel(const float* __restrict__ in,
                                                    u16* __restrict__ o) {
  int i = blockIdx.x * 256 + threadIdx.x;  // over H*C*D
  const int d = i % kD;
  const int c = (i / kD) % kC;
  const int h = i / (kD * kC);
  o[((size_t)h * kD + d) * kC + c] = f2bf(in[i]);
}

// Wo [HD][C] fp32 -> WoT [C][HD] bf16
__global__ __launch_bounds__(256) void cvt_wo_kernel(const float* __restrict__ in,
                                                     u16* __restrict__ o) {
  int i = blockIdx.x * 256 + threadIdx.x;  // i = j*C + c
  const int c = i & (kC - 1);
  const int j = i >> 10;
  o[(size_t)c * (kH * kD) + j] = f2bf(in[i]);
}

// ---------------- QKV projection GEMM ----------------
// Per wave: one 32x64 tile of q/k/v for one (b, mat, h).  M=T, N=D=64, K=C=1024.
// 8 WMMAs per K-step share 4 B-fragments; K-loop software-pipelined.
__global__ __launch_bounds__(256) void qkv_gemm_kernel(
    const u16* __restrict__ xb, const u16* __restrict__ wT,
    u16* __restrict__ q, u16* __restrict__ k, u16* __restrict__ vT) {
  const int lane = threadIdx.x & 31;
  const int wave = threadIdx.x >> 5;
  int gt = blockIdx.x * 8 + wave;            // 12288 wave tiles total
  const int mtile = gt & 63;  gt >>= 6;      // T/32 = 64
  const int h = gt & 15;      gt >>= 4;
  const int mat = gt % 3;                    // 0=q 1=k 2=v
  const int b = gt / 3;
  const int m0 = mtile * 32;
  const int col = lane & 15;
  const int hi = lane >> 4;

  const u16* A0 = xb + ((size_t)b * kT + m0 + col) * kC;
  const u16* A1 = A0 + (size_t)16 * kC;
  const u16* W = wT + (size_t)mat * WN + (size_t)h * kD * kC;
  const u16* Br0 = W + (size_t)(0  + col) * kC;
  const u16* Br1 = W + (size_t)(16 + col) * kC;
  const u16* Br2 = W + (size_t)(32 + col) * kC;
  const u16* Br3 = W + (size_t)(48 + col) * kC;

  v8f acc[2][4] = {};
  // prologue: k-step 0 fragments
  v16bf a0 = load_afrag(A0, lane), a1 = load_afrag(A1, lane);
  v16bf b0f = load_bfrag(Br0, lane), b1f = load_bfrag(Br1, lane);
  v16bf b2f = load_bfrag(Br2, lane), b3f = load_bfrag(Br3, lane);

#pragma unroll 2
  for (int k0 = 32; k0 <= kC; k0 += 32) {
    const int kn = (k0 < kC) ? k0 : 0;  // last iter: harmless reload of step 0
    __builtin_prefetch(A0 + kn + 256, 0, 3);
    // issue next k-step's loads BEFORE consuming current fragments
    v16bf na0 = load_afrag(A0 + kn, lane);
    v16bf na1 = load_afrag(A1 + kn, lane);
    v16bf nb0 = load_bfrag(Br0 + kn, lane);
    v16bf nb1 = load_bfrag(Br1 + kn, lane);
    v16bf nb2 = load_bfrag(Br2 + kn, lane);
    v16bf nb3 = load_bfrag(Br3 + kn, lane);
    acc[0][0] = wmma_bf16(a0, b0f, acc[0][0]);
    acc[0][1] = wmma_bf16(a0, b1f, acc[0][1]);
    acc[0][2] = wmma_bf16(a0, b2f, acc[0][2]);
    acc[0][3] = wmma_bf16(a0, b3f, acc[0][3]);
    acc[1][0] = wmma_bf16(a1, b0f, acc[1][0]);
    acc[1][1] = wmma_bf16(a1, b1f, acc[1][1]);
    acc[1][2] = wmma_bf16(a1, b2f, acc[1][2]);
    acc[1][3] = wmma_bf16(a1, b3f, acc[1][3]);
    a0 = na0; a1 = na1; b0f = nb0; b1f = nb1; b2f = nb2; b3f = nb3;
  }

  const size_t bh = (size_t)b * kH + h;
#pragma unroll
  for (int mt = 0; mt < 2; ++mt) {
#pragma unroll
    for (int nt = 0; nt < 4; ++nt) {
#pragma unroll
      for (int r = 0; r < 8; ++r) {
        const int row = m0 + mt * 16 + hi * 8 + r;
        const int c = nt * 16 + col;
        const u16 val = f2bf(acc[mt][nt][r]);
        if (mat == 0)      q[(bh * kT + row) * kD + c] = val;          // [B,H,T,D]
        else if (mat == 1) k[(bh * kT + row) * kD + c] = val;          // [B,H,T,D]
        else               vT[(bh * kD + c) * (size_t)kT + row] = val; // [B,H,D,T]
      }
    }
  }
}

// ---------------- flash attention ----------------
// 128 query rows per block, 4 waves; wave w owns 32 rows (2 M-tiles).
// Per 64-key block: 16 WMMAs (QK^T) -> LDS -> full-wave online softmax ->
// P (bf16) in LDS -> 16 WMMAs (PV) with accumulator rescale.
__global__ __launch_bounds__(128) void attn_kernel(
    const u16* __restrict__ q, const u16* __restrict__ k,
    const u16* __restrict__ vT, u16* __restrict__ att) {
  __shared__ float lds_s[4][32][64];
  __shared__ u16   lds_p[4][32][64];
  __shared__ float row_scale[4][32];
  __shared__ float row_l[4][32];

  const int lane = threadIdx.x & 31;
  const int w    = threadIdx.x >> 5;
  int blk = blockIdx.x;                 // B*H*(T/128) = 1024
  const int qblk = blk & 15; blk >>= 4; // T/128 = 16
  const int h = blk & 15;
  const int b = blk >> 4;
  const size_t bh = (size_t)b * kH + h;
  const int t0w = qblk * 128 + w * 32;  // this wave's first query row
  const int col = lane & 15;
  const int hi = lane >> 4;
  const float scale = 0.125f;           // 1/sqrt(64)
  const float NEG = -3.0e38f;

  // q fragments for both 16-row tiles (K = d, two 32-chunks)
  const u16* Aq0 = q + (bh * kT + t0w + col) * kD;
  const u16* Aq1 = Aq0 + (size_t)16 * kD;
  const v16bf qa00 = load_afrag(Aq0, lane),      qa01 = load_afrag(Aq0 + 32, lane);
  const v16bf qa10 = load_afrag(Aq1, lane),      qa11 = load_afrag(Aq1 + 32, lane);

  v8f o[2][4] = {};
  float m_i = NEG, l_i = 0.f;

  const int nblocks = 2 * qblk + 2;  // causal: 64-key blocks covering rows [0, t0blk+128)
  for (int sb = 0; sb < nblocks; ++sb) {
    const int s0 = sb * 64;
    // ---- S = q . k^T : pipeline over the 4 key sub-tiles ----
    const u16* Bkb = k + (bh * kT + s0 + col) * kD;
    v16bf cb0 = load_bfrag(Bkb, lane);
    v16bf cb1 = load_bfrag(Bkb + 32, lane);
#pragma unroll
    for (int nt = 0; nt < 4; ++nt) {
      const int ntn = (nt < 3) ? nt + 1 : nt;
      v16bf nb0 = load_bfrag(Bkb + (size_t)ntn * 16 * kD, lane);
      v16bf nb1 = load_bfrag(Bkb + (size_t)ntn * 16 * kD + 32, lane);
      v8f s0a = {}, s1a = {};
      s0a = wmma_bf16(qa00, cb0, s0a);
      s0a = wmma_bf16(qa01, cb1, s0a);
      s1a = wmma_bf16(qa10, cb0, s1a);
      s1a = wmma_bf16(qa11, cb1, s1a);
#pragma unroll
      for (int r = 0; r < 8; ++r) {
        lds_s[w][hi * 8 + r][nt * 16 + col]      = s0a[r];
        lds_s[w][16 + hi * 8 + r][nt * 16 + col] = s1a[r];
      }
      cb0 = nb0; cb1 = nb1;
    }
    __syncthreads();
    // ---- online softmax: all 32 lanes, lane = query row ----
    {
      const int ig = t0w + lane;
      float mnew = m_i;
      for (int c = 0; c < 64; ++c) {
        const float sv = (s0 + c <= ig) ? lds_s[w][lane][c] * scale : NEG;
        mnew = fmaxf(mnew, sv);
      }
      const float corr = __expf(m_i - mnew);
      float ls = l_i * corr;
      for (int c = 0; c < 64; ++c) {
        const float sv = (s0 + c <= ig) ? lds_s[w][lane][c] * scale : NEG;
        const float p = __expf(sv - mnew);
        lds_p[w][lane][c] = f2bf(p);
        ls += p;
      }
      m_i = mnew; l_i = ls;
      row_scale[w][lane] = corr;
    }
    __syncthreads();
    // ---- rescale accumulators, then O += P . V (pipelined over nt) ----
    float cs0[8], cs1[8];
#pragma unroll
    for (int r = 0; r < 8; ++r) {
      cs0[r] = row_scale[w][hi * 8 + r];
      cs1[r] = row_scale[w][16 + hi * 8 + r];
    }
#pragma unroll
    for (int nt = 0; nt < 4; ++nt)
#pragma unroll
      for (int r = 0; r < 8; ++r) { o[0][nt][r] *= cs0[r]; o[1][nt][r] *= cs1[r]; }

    const v16bf pa00 = load_afrag(&lds_p[w][col][0],       lane);
    const v16bf pa01 = load_afrag(&lds_p[w][col][32],      lane);
    const v16bf pa10 = load_afrag(&lds_p[w][16 + col][0],  lane);
    const v16bf pa11 = load_afrag(&lds_p[w][16 + col][32], lane);

    const u16* Bvb = vT + (bh * kD + col) * (size_t)kT + s0;  // row d, contiguous s
    v16bf vb0 = load_bfrag(Bvb, lane);
    v16bf vb1 = load_bfrag(Bvb + 32, lane);
#pragma unroll
    for (int nt = 0; nt < 4; ++nt) {
      const int ntn = (nt < 3) ? nt + 1 : nt;
      v16bf nv0 = load_bfrag(Bvb + (size_t)ntn * 16 * kT, lane);
      v16bf nv1 = load_bfrag(Bvb + (size_t)ntn * 16 * kT + 32, lane);
      o[0][nt] = wmma_bf16(pa00, vb0, o[0][nt]);
      o[0][nt] = wmma_bf16(pa01, vb1, o[0][nt]);
      o[1][nt] = wmma_bf16(pa10, vb0, o[1][nt]);
      o[1][nt] = wmma_bf16(pa11, vb1, o[1][nt]);
      vb0 = nv0; vb1 = nv1;
    }
    __syncthreads();
  }

  row_l[w][lane] = l_i;
  __syncthreads();
  float inv0[8], inv1[8];
#pragma unroll
  for (int r = 0; r < 8; ++r) {
    inv0[r] = 1.0f / row_l[w][hi * 8 + r];
    inv1[r] = 1.0f / row_l[w][16 + hi * 8 + r];
  }

  // att layout [B, T, H*D] bf16 (row-major => A operand of final projection)
  u16* outp = att + ((size_t)b * kT + t0w) * (kH * kD) + h * kD;
#pragma unroll
  for (int nt = 0; nt < 4; ++nt) {
#pragma unroll
    for (int r = 0; r < 8; ++r) {
      const size_t ro0 = (size_t)(hi * 8 + r) * (kH * kD);
      const size_t ro1 = (size_t)(16 + hi * 8 + r) * (kH * kD);
      outp[ro0 + nt * 16 + col] = f2bf(o[0][nt][r] * inv0[r]);
      outp[ro1 + nt * 16 + col] = f2bf(o[1][nt][r] * inv1[r]);
    }
  }
}

// ---------------- output projection ----------------
// out[m, n] = att[m, :] . Wo[:, n] + bo[n];  M = B*T = 8192, N = C, K = H*D.
// 32x64 tile per wave, software-pipelined K-loop.
__global__ __launch_bounds__(256) void proj_kernel(
    const u16* __restrict__ att, const u16* __restrict__ woT,
    const float* __restrict__ bo, float* __restrict__ out) {
  const int lane = threadIdx.x & 31;
  const int wave = threadIdx.x >> 5;
  int gt = blockIdx.x * 8 + wave;   // 4096 wave tiles
  const int ntile = gt & 15;        // C/64 = 16
  const int mtile = gt >> 4;        // (B*T)/32 = 256
  const int m0 = mtile * 32;
  const int n0 = ntile * 64;
  const int col = lane & 15;
  const int hi = lane >> 4;

  const u16* A0 = att + ((size_t)m0 + col) * kC;
  const u16* A1 = A0 + (size_t)16 * kC;
  const u16* Br0 = woT + (size_t)(n0 + 0  + col) * kC;
  const u16* Br1 = woT + (size_t)(n0 + 16 + col) * kC;
  const u16* Br2 = woT + (size_t)(n0 + 32 + col) * kC;
  const u16* Br3 = woT + (size_t)(n0 + 48 + col) * kC;

  v8f acc[2][4] = {};
  v16bf a0 = load_afrag(A0, lane), a1 = load_afrag(A1, lane);
  v16bf b0f = load_bfrag(Br0, lane), b1f = load_bfrag(Br1, lane);
  v16bf b2f = load_bfrag(Br2, lane), b3f = load_bfrag(Br3, lane);

#pragma unroll 2
  for (int k0 = 32; k0 <= kC; k0 += 32) {
    const int kn = (k0 < kC) ? k0 : 0;
    __builtin_prefetch(A0 + kn + 256, 0, 3);
    v16bf na0 = load_afrag(A0 + kn, lane);
    v16bf na1 = load_afrag(A1 + kn, lane);
    v16bf nb0 = load_bfrag(Br0 + kn, lane);
    v16bf nb1 = load_bfrag(Br1 + kn, lane);
    v16bf nb2 = load_bfrag(Br2 + kn, lane);
    v16bf nb3 = load_bfrag(Br3 + kn, lane);
    acc[0][0] = wmma_bf16(a0, b0f, acc[0][0]);
    acc[0][1] = wmma_bf16(a0, b1f, acc[0][1]);
    acc[0][2] = wmma_bf16(a0, b2f, acc[0][2]);
    acc[0][3] = wmma_bf16(a0, b3f, acc[0][3]);
    acc[1][0] = wmma_bf16(a1, b0f, acc[1][0]);
    acc[1][1] = wmma_bf16(a1, b1f, acc[1][1]);
    acc[1][2] = wmma_bf16(a1, b2f, acc[1][2]);
    acc[1][3] = wmma_bf16(a1, b3f, acc[1][3]);
    a0 = na0; a1 = na1; b0f = nb0; b1f = nb1; b2f = nb2; b3f = nb3;
  }

  const float bb[4] = { bo[n0 + 0  + col], bo[n0 + 16 + col],
                        bo[n0 + 32 + col], bo[n0 + 48 + col] };
#pragma unroll
  for (int mt = 0; mt < 2; ++mt) {
#pragma unroll
    for (int nt = 0; nt < 4; ++nt) {
#pragma unroll
      for (int r = 0; r < 8; ++r) {
        const size_t row = (size_t)(m0 + mt * 16 + hi * 8 + r);
        out[row * kC + n0 + nt * 16 + col] = acc[mt][nt][r] + bb[nt];
      }
    }
  }
}

// ---------------- launch ----------------

extern "C" void kernel_launch(void* const* d_in, const int* in_sizes, int n_in,
                              void* d_out, int out_size, void* d_ws, size_t ws_size,
                              hipStream_t stream) {
  (void)in_sizes; (void)n_in; (void)out_size; (void)ws_size;
  const float* x  = (const float*)d_in[0];
  const float* Wq = (const float*)d_in[1];
  const float* Wk = (const float*)d_in[2];
  const float* Wv = (const float*)d_in[3];
  const float* Wo = (const float*)d_in[4];
  const float* bo = (const float*)d_in[5];
  float* out = (float*)d_out;

  // workspace layout (bf16 as u16), ~88 MB total
  u16* ws   = (u16*)d_ws;
  u16* xb   = ws;                 // XN
  u16* wT   = xb + XN;            // 3*WN  (WqT, WkT, WvT contiguous, [H][D][C])
  u16* woT  = wT + 3 * WN;        // WN    ([C][H*D])
  u16* qb   = woT + WN;           // QN    ([B,H,T,D])
  u16* kb   = qb + QN;            // QN    ([B,H,T,D])
  u16* vtb  = kb + QN;            // QN    ([B,H,D,T])
  u16* attb = vtb + QN;           // XN    ([B,T,H*D])

  cvt_x_kernel  <<<(int)(XN / 256), 256, 0, stream>>>(x, xb, (int)XN);
  cvt_w_kernel  <<<(int)(WN / 256), 256, 0, stream>>>(Wq, wT);
  cvt_w_kernel  <<<(int)(WN / 256), 256, 0, stream>>>(Wk, wT + WN);
  cvt_w_kernel  <<<(int)(WN / 256), 256, 0, stream>>>(Wv, wT + 2 * WN);
  cvt_wo_kernel <<<(int)(WN / 256), 256, 0, stream>>>(Wo, woT);

  qkv_gemm_kernel<<<1536, 256, 0, stream>>>(xb, wT, qb, kb, vtb);
  attn_kernel    <<<1024, 128, 0, stream>>>(qb, kb, vtb, attb);
  proj_kernel    <<<512,  256, 0, stream>>>(attb, woT, bo, out);
}
